// DDiTBlock_32495722562197
// MI455X (gfx1250) — matrix operand order repaired and checked by reading, not compiled, not run on hardware
//
#include <hip/hip_runtime.h>
#include <hip/hip_bf16.h>

// ---------------------------------------------------------------------------
// DiT block for MI455X (gfx1250, wave32, WMMA).
// All five GEMMs + both attention matmuls run on v_wmma_f32_16x16x32_bf16
// (f32 accumulate). Weights fit in the 192MB L2 as bf16, so fragments are
// loaded straight from global memory as 16-byte chunks per the ISA VGPR
// layouts. Each GEMM wave computes a 32x64 tile (2 A-frags x 4 B-frags,
// 8 WMMAs per 12 b128 loads) so weight B-fragments are reused twice,
// doubling FLOP/byte vs a 16x64 strip. Attention is single-pass flash-style
// with shfl row reductions and an LDS transpose of P.
// ---------------------------------------------------------------------------

typedef __bf16 bf16_t;
typedef __attribute__((ext_vector_type(16))) __bf16 v16bf;
typedef __attribute__((ext_vector_type(8)))  __bf16 v8bf;
typedef __attribute__((ext_vector_type(8)))  float  v8f;

#define B_  2
#define S_  2048
#define D_  1024
#define H_  16
#define DH_ 64
#define MS_ 4096              // B*S rows
#define ADA_N 6144

__device__ inline v8f zero8() {
  v8f z;
  #pragma unroll
  for (int i = 0; i < 8; ++i) z[i] = 0.f;
  return z;
}

__device__ inline v16bf join16(v8bf lo, v8bf hi) {
  return __builtin_shufflevector(lo, hi, 0,1,2,3,4,5,6,7,8,9,10,11,12,13,14,15);
}

// A fragment (16x32 bf16): row-major src, row stride ld. ISA layout:
// lane<16 : elems 0..7 = K k0+0..7,  elems 8..15 = K k0+16..23
// lane>=16: elems 0..7 = K k0+8..15, elems 8..15 = K k0+24..31
__device__ inline v16bf load_a_frag(const bf16_t* src, int ld, int row0, int k0, int lane) {
  const bf16_t* p = src + (size_t)(row0 + (lane & 15)) * ld + k0 + ((lane >> 4) << 3);
  v8bf lo = *(const v8bf*)(p);
  v8bf hi = *(const v8bf*)(p + 16);
  return join16(lo, hi);
}

// B fragment (32x16): B[k,n] = W[n0+n][k0+k], W row-major stride ld.
// lane<16: elems 0..7 = K k0+0..7, 8..15 = K k0+16..23; lane>=16: K k0+8..15, k0+24..31.
__device__ inline v16bf load_b_frag(const bf16_t* W, int ld, int n0, int k0, int lane) {
  const bf16_t* p = W + (size_t)(n0 + (lane & 15)) * ld + k0 + ((lane >> 4) << 3);
  v8bf lo = *(const v8bf*)(p);
  v8bf hi = *(const v8bf*)(p + 16);
  return join16(lo, hi);
}

__device__ inline v8f wmma_bf16(v16bf a, v16bf b, v8f c) {
  return __builtin_amdgcn_wmma_f32_16x16x32_bf16(false, a, false, b, (short)0, c, false, false);
}

// One wave computes a 32(M) x 64(N) tile of C = A[M,K] @ W[N,K]^T.
// B-fragments (weights) are reused across both A row-fragments.
__device__ inline void gemm_tile32(const bf16_t* A, const bf16_t* W, int K,
                                   int m0, int n0, int lane, v8f acc[2][4]) {
  #pragma unroll
  for (int mi = 0; mi < 2; ++mi)
    #pragma unroll
    for (int nb = 0; nb < 4; ++nb) acc[mi][nb] = zero8();
  for (int k0 = 0; k0 < K; k0 += 32) {
    v16bf a0 = load_a_frag(A, K, m0,      k0, lane);
    v16bf a1 = load_a_frag(A, K, m0 + 16, k0, lane);
    #pragma unroll
    for (int nb = 0; nb < 4; ++nb) {
      v16bf b = load_b_frag(W, K, n0 + nb * 16, k0, lane);
      acc[0][nb] = wmma_bf16(a0, b, acc[0][nb]);
      acc[1][nb] = wmma_bf16(a1, b, acc[1][nb]);
    }
  }
}

// ----------------------------- utility kernels -----------------------------

__global__ __launch_bounds__(256) void k_f32_to_bf16(const float* in, bf16_t* out, int n) {
  int i = blockIdx.x * 256 + threadIdx.x;
  if (i < n) out[i] = (bf16_t)in[i];
}

__device__ inline float block_sum(float v, float* red) {
  #pragma unroll
  for (int off = 16; off; off >>= 1) v += __shfl_xor(v, off, 32);
  int wid = threadIdx.x >> 5;
  if ((threadIdx.x & 31) == 0) red[wid] = v;
  __syncthreads();
  float s = 0.f;
  #pragma unroll
  for (int i = 0; i < 8; ++i) s += red[i];
  __syncthreads();
  return s;
}

// LayerNorm + modulate -> bf16 (shiftOff/scaleOff select ada chunks)
__global__ __launch_bounds__(256)
void k_ln_mod(const float* xin, const float* w, const float* ada,
              int shiftOff, int scaleOff, bf16_t* out) {
  __shared__ float red[8];
  int row = blockIdx.x, tid = threadIdx.x;
  const float* xr = xin + (size_t)row * D_;
  float v[4], s = 0.f;
  #pragma unroll
  for (int i = 0; i < 4; ++i) { v[i] = xr[tid + i * 256]; s += v[i]; }
  float mu = block_sum(s, red) * (1.f / D_);
  float sq = 0.f;
  #pragma unroll
  for (int i = 0; i < 4; ++i) { float d = v[i] - mu; sq += d * d; }
  float var = block_sum(sq, red) * (1.f / D_);
  float rstd = rsqrtf(var + 1e-5f);
  const float* arow = ada + (size_t)row * ADA_N;
  #pragma unroll
  for (int i = 0; i < 4; ++i) {
    int col = tid + i * 256;
    float h = (v[i] - mu) * rstd * w[col];
    float sc = arow[scaleOff + col], sh = arow[shiftOff + col];
    out[(size_t)row * D_ + col] = (bf16_t)(h * (1.f + sc) + sh);
  }
}

// RoPE over q,k,v (reference rotates all three). Writes q,k [B,H,S,64] bf16
// and v transposed [B,H,64,S] bf16 so attention B-fragments are contiguous-K.
__global__ __launch_bounds__(256)
void k_rope(const float* qkv, const float* cosT, const float* sinT,
            bf16_t* q, bf16_t* k, bf16_t* vT) {
  int idx = blockIdx.x * 256 + threadIdx.x;     // B*S*3*H*32 pairs
  if (idx >= B_ * S_ * 3 * H_ * 32) return;
  int j = idx & 31; int t = idx >> 5;
  int h = t & 15;  t >>= 4;
  int w = t % 3;   t /= 3;
  int s = t & (S_ - 1); int b = t >> 11;
  size_t base = ((size_t)(b * S_ + s) * 3 + w) * (size_t)D_ + h * DH_;
  float x1 = qkv[base + j], x2 = qkv[base + j + 32];
  float c1 = cosT[s * DH_ + j],      s1 = sinT[s * DH_ + j];
  float c2 = cosT[s * DH_ + j + 32], s2 = sinT[s * DH_ + j + 32];
  float o1 = x1 * c1 - x2 * s1;
  float o2 = x2 * c2 + x1 * s2;
  if (w == 2) {
    size_t vb = (size_t)(b * H_ + h) * DH_;
    vT[(vb + j) * S_ + s]      = (bf16_t)o1;
    vT[(vb + j + 32) * S_ + s] = (bf16_t)o2;
  } else {
    bf16_t* dst = (w == 0) ? q : k;
    size_t ob = (size_t)((b * H_ + h) * S_ + s) * DH_;
    dst[ob + j]      = (bf16_t)o1;
    dst[ob + j + 32] = (bf16_t)o2;
  }
}

// ------------------------------- GEMM kernels ------------------------------

__global__ __launch_bounds__(32)
void k_gemm_ada(const bf16_t* A, const bf16_t* W, const float* bias, float* out) {
  int lane = threadIdx.x, n0 = blockIdx.x * 64, m0 = blockIdx.y * 32;
  v8f acc[2][4]; gemm_tile32(A, W, D_, m0, n0, lane, acc);
  int cl = lane & 15;
  #pragma unroll
  for (int mi = 0; mi < 2; ++mi) {
    int rbase = m0 + mi * 16 + ((lane >> 4) << 3);
    #pragma unroll
    for (int nb = 0; nb < 4; ++nb) {
      int col = n0 + nb * 16 + cl; float bv = bias[col];
      #pragma unroll
      for (int r = 0; r < 8; ++r)
        out[(size_t)(rbase + r) * ADA_N + col] = acc[mi][nb][r] + bv;
    }
  }
}

__global__ __launch_bounds__(32)
void k_gemm_qkv(const bf16_t* A, const bf16_t* W, float* out) {
  int lane = threadIdx.x, n0 = blockIdx.x * 64, m0 = blockIdx.y * 32;
  v8f acc[2][4]; gemm_tile32(A, W, D_, m0, n0, lane, acc);
  int cl = lane & 15;
  #pragma unroll
  for (int mi = 0; mi < 2; ++mi) {
    int rbase = m0 + mi * 16 + ((lane >> 4) << 3);
    #pragma unroll
    for (int nb = 0; nb < 4; ++nb) {
      int col = n0 + nb * 16 + cl;
      #pragma unroll
      for (int r = 0; r < 8; ++r)
        out[(size_t)(rbase + r) * (3 * D_) + col] = acc[mi][nb][r];
    }
  }
}

// x2 = x + gate_msa * (attn @ Wout^T)
__global__ __launch_bounds__(32)
void k_gemm_outproj(const bf16_t* A, const bf16_t* W, const float* x,
                    const float* ada, float* x2) {
  int lane = threadIdx.x, n0 = blockIdx.x * 64, m0 = blockIdx.y * 32;
  v8f acc[2][4]; gemm_tile32(A, W, D_, m0, n0, lane, acc);
  int cl = lane & 15;
  #pragma unroll
  for (int mi = 0; mi < 2; ++mi) {
    int rbase = m0 + mi * 16 + ((lane >> 4) << 3);
    #pragma unroll
    for (int nb = 0; nb < 4; ++nb) {
      int col = n0 + nb * 16 + cl;
      #pragma unroll
      for (int r = 0; r < 8; ++r) {
        size_t row = rbase + r;
        float g = ada[row * ADA_N + 2 * D_ + col];
        x2[row * D_ + col] = x[row * D_ + col] + g * acc[mi][nb][r];
      }
    }
  }
}

// mlp_h = gelu_tanh(h2 @ Wmlp1^T + b) as bf16
__global__ __launch_bounds__(32)
void k_gemm_mlp1(const bf16_t* A, const bf16_t* W, const float* bias, bf16_t* out) {
  int lane = threadIdx.x, n0 = blockIdx.x * 64, m0 = blockIdx.y * 32;
  v8f acc[2][4]; gemm_tile32(A, W, D_, m0, n0, lane, acc);
  int cl = lane & 15;
  #pragma unroll
  for (int mi = 0; mi < 2; ++mi) {
    int rbase = m0 + mi * 16 + ((lane >> 4) << 3);
    #pragma unroll
    for (int nb = 0; nb < 4; ++nb) {
      int col = n0 + nb * 16 + cl; float bv = bias[col];
      #pragma unroll
      for (int r = 0; r < 8; ++r) {
        float u = acc[mi][nb][r] + bv;
        float t = tanhf(0.7978845608028654f * (u + 0.044715f * u * u * u));
        out[(size_t)(rbase + r) * (4 * D_) + col] = (bf16_t)(0.5f * u * (1.f + t));
      }
    }
  }
}

// out = x2 + gate_mlp * (mlp_h @ Wmlp2^T + b)
__global__ __launch_bounds__(32)
void k_gemm_mlp2(const bf16_t* A, const bf16_t* W, const float* bias,
                 const float* x2, const float* ada, float* out) {
  int lane = threadIdx.x, n0 = blockIdx.x * 64, m0 = blockIdx.y * 32;
  v8f acc[2][4]; gemm_tile32(A, W, 4 * D_, m0, n0, lane, acc);
  int cl = lane & 15;
  #pragma unroll
  for (int mi = 0; mi < 2; ++mi) {
    int rbase = m0 + mi * 16 + ((lane >> 4) << 3);
    #pragma unroll
    for (int nb = 0; nb < 4; ++nb) {
      int col = n0 + nb * 16 + cl; float bv = bias[col];
      #pragma unroll
      for (int r = 0; r < 8; ++r) {
        size_t row = rbase + r;
        float g = ada[row * ADA_N + 5 * D_ + col];
        out[row * D_ + col] = x2[row * D_ + col] + g * (acc[mi][nb][r] + bv);
      }
    }
  }
}

// --------------------------- flash attention -------------------------------
// One wave per (b, h, 16-row q tile). Key tiles of 32. Mask uses finite -1e9
// so fully-masked rows degenerate to the reference's uniform softmax.
__global__ __launch_bounds__(32)
void k_attn(const bf16_t* q, const bf16_t* k, const bf16_t* vT,
            const unsigned char* mask, bf16_t* attnOut) {
  __shared__ bf16_t pLds[16 * 32];
  int lane = threadIdx.x;
  int bid = blockIdx.x;                 // B*H*(S/16)
  int sTile = bid & 127; bid >>= 7;
  int h = bid & 15; int b = bid >> 4;
  int s0 = sTile * 16;
  const bf16_t* qh = q  + (size_t)(b * H_ + h) * S_ * DH_;
  const bf16_t* kh = k  + (size_t)(b * H_ + h) * S_ * DH_;
  const bf16_t* vh = vT + (size_t)(b * H_ + h) * DH_ * S_;
  const unsigned char* mb = mask + (size_t)b * S_ * S_;

  v16bf qa0 = load_a_frag(qh, DH_, s0, 0, lane);
  v16bf qa1 = load_a_frag(qh, DH_, s0, 32, lane);

  int rbase = (lane >> 4) << 3;         // row offset of this half's C rows
  int cl = lane & 15;

  float m[8], lsum[8]; v8f acc[4];
  #pragma unroll
  for (int r = 0; r < 8; ++r) { m[r] = -1e30f; lsum[r] = 0.f; }
  #pragma unroll
  for (int i = 0; i < 4; ++i) acc[i] = zero8();

  for (int t0 = 0; t0 < S_; t0 += 32) {
    v8f s0f = zero8(), s1f = zero8();
    s0f = wmma_bf16(qa0, load_b_frag(kh, DH_, t0,       0,  lane), s0f);
    s0f = wmma_bf16(qa1, load_b_frag(kh, DH_, t0,       32, lane), s0f);
    s1f = wmma_bf16(qa0, load_b_frag(kh, DH_, t0 + 16,  0,  lane), s1f);
    s1f = wmma_bf16(qa1, load_b_frag(kh, DH_, t0 + 16,  32, lane), s1f);
    #pragma unroll
    for (int r = 0; r < 8; ++r) {
      int srow = s0 + rbase + r;
      const unsigned char* mp = mb + (size_t)srow * S_ + t0 + cl;
      float v0 = mp[0]  ? -1e9f : s0f[r] * 0.125f;
      float v1 = mp[16] ? -1e9f : s1f[r] * 0.125f;
      float tmax = fmaxf(v0, v1);
      #pragma unroll
      for (int off = 1; off < 16; off <<= 1) tmax = fmaxf(tmax, __shfl_xor(tmax, off, 32));
      float mn = fmaxf(m[r], tmax);
      float corr = __expf(m[r] - mn);
      float p0 = __expf(v0 - mn), p1 = __expf(v1 - mn);
      float ps = p0 + p1;
      #pragma unroll
      for (int off = 1; off < 16; off <<= 1) ps += __shfl_xor(ps, off, 32);
      lsum[r] = lsum[r] * corr + ps;
      m[r] = mn;
      #pragma unroll
      for (int db = 0; db < 4; ++db) acc[db][r] *= corr;
      pLds[(rbase + r) * 32 + cl]      = (bf16_t)p0;
      pLds[(rbase + r) * 32 + cl + 16] = (bf16_t)p1;
    }
    // single wave: LDS write->read is in-order; read P back as an A fragment
    v16bf pa = load_a_frag(pLds, 32, 0, 0, lane);
    #pragma unroll
    for (int db = 0; db < 4; ++db) {
      v16bf vb = load_b_frag(vh, S_, db * 16, t0, lane);
      acc[db] = wmma_bf16(pa, vb, acc[db]);
    }
  }
  #pragma unroll
  for (int db = 0; db < 4; ++db) {
    #pragma unroll
    for (int r = 0; r < 8; ++r) {
      float o = acc[db][r] / lsum[r];
      attnOut[(size_t)(b * S_ + s0 + rbase + r) * D_ + h * DH_ + db * 16 + cl] = (bf16_t)o;
    }
  }
}

// ------------------------------- launcher ----------------------------------

extern "C" void kernel_launch(void* const* d_in, const int* in_sizes, int n_in,
                              void* d_out, int out_size, void* d_ws, size_t ws_size,
                              hipStream_t stream) {
  const float* x      = (const float*)d_in[0];
  const float* cosT   = (const float*)d_in[1];
  const float* sinT   = (const float*)d_in[2];
  const float* c      = (const float*)d_in[3];
  const float* w_n1   = (const float*)d_in[4];
  const float* w_qkv  = (const float*)d_in[5];
  const float* w_out  = (const float*)d_in[6];
  const float* w_n2   = (const float*)d_in[7];
  const float* w_mlp1 = (const float*)d_in[8];
  const float* b_mlp1 = (const float*)d_in[9];
  const float* w_mlp2 = (const float*)d_in[10];
  const float* b_mlp2 = (const float*)d_in[11];
  const float* w_ada  = (const float*)d_in[12];
  const float* b_ada  = (const float*)d_in[13];
  const unsigned char* mask = (const unsigned char*)d_in[14];
  float* out = (float*)d_out;

  char* wp = (char*)d_ws;
  auto alloc = [&](size_t bytes) -> void* {
    void* p = (void*)wp; wp += (bytes + 255) & ~(size_t)255; return p;
  };
  bf16_t* c_bf     = (bf16_t*)alloc((size_t)MS_ * D_ * 2);
  bf16_t* wada_bf  = (bf16_t*)alloc((size_t)ADA_N * D_ * 2);
  bf16_t* wqkv_bf  = (bf16_t*)alloc((size_t)3 * D_ * D_ * 2);
  bf16_t* wout_bf  = (bf16_t*)alloc((size_t)D_ * D_ * 2);
  bf16_t* wmlp1_bf = (bf16_t*)alloc((size_t)4 * D_ * D_ * 2);
  bf16_t* wmlp2_bf = (bf16_t*)alloc((size_t)D_ * 4 * D_ * 2);
  float*  ada      = (float*) alloc((size_t)MS_ * ADA_N * 4);
  bf16_t* h1       = (bf16_t*)alloc((size_t)MS_ * D_ * 2);
  float*  qkv_f    = (float*) alloc((size_t)MS_ * 3 * D_ * 4);
  bf16_t* q_bf     = (bf16_t*)alloc((size_t)MS_ * D_ * 2);
  bf16_t* k_bf     = (bf16_t*)alloc((size_t)MS_ * D_ * 2);
  bf16_t* vT_bf    = (bf16_t*)alloc((size_t)MS_ * D_ * 2);
  bf16_t* attn_bf  = (bf16_t*)alloc((size_t)MS_ * D_ * 2);
  float*  x2       = (float*) alloc((size_t)MS_ * D_ * 4);
  bf16_t* h2       = (bf16_t*)alloc((size_t)MS_ * D_ * 2);
  bf16_t* mlp_h    = (bf16_t*)alloc((size_t)MS_ * 4 * D_ * 2);

  auto conv = [&](const float* in, bf16_t* o, size_t n) {
    k_f32_to_bf16<<<dim3((unsigned)((n + 255) / 256)), dim3(256), 0, stream>>>(in, o, (int)n);
  };
  conv(c,      c_bf,     (size_t)MS_ * D_);
  conv(w_ada,  wada_bf,  (size_t)ADA_N * D_);
  conv(w_qkv,  wqkv_bf,  (size_t)3 * D_ * D_);
  conv(w_out,  wout_bf,  (size_t)D_ * D_);
  conv(w_mlp1, wmlp1_bf, (size_t)4 * D_ * D_);
  conv(w_mlp2, wmlp2_bf, (size_t)D_ * 4 * D_);

  // ada = c @ w_ada^T + b_ada
  k_gemm_ada<<<dim3(ADA_N / 64, MS_ / 32), dim3(32), 0, stream>>>(c_bf, wada_bf, b_ada, ada);
  // h1 = modulate(LN(x, w_n1), shift_msa, scale_msa)
  k_ln_mod<<<dim3(MS_), dim3(256), 0, stream>>>(x, w_n1, ada, 0, D_, h1);
  // qkv = h1 @ w_qkv^T
  k_gemm_qkv<<<dim3(3 * D_ / 64, MS_ / 32), dim3(32), 0, stream>>>(h1, wqkv_bf, qkv_f);
  // RoPE (q, k, v) + layout for attention
  {
    int n = B_ * S_ * 3 * H_ * 32;
    k_rope<<<dim3((n + 255) / 256), dim3(256), 0, stream>>>(qkv_f, cosT, sinT, q_bf, k_bf, vT_bf);
  }
  // flash attention
  k_attn<<<dim3(B_ * H_ * (S_ / 16)), dim3(32), 0, stream>>>(q_bf, k_bf, vT_bf, mask, attn_bf);
  // x2 = x + gate_msa * (attn @ w_out^T)
  k_gemm_outproj<<<dim3(D_ / 64, MS_ / 32), dim3(32), 0, stream>>>(attn_bf, wout_bf, x, ada, x2);
  // h2 = modulate(LN(x2, w_n2), shift_mlp, scale_mlp)
  k_ln_mod<<<dim3(MS_), dim3(256), 0, stream>>>(x2, w_n2, ada, 3 * D_, 4 * D_, h2);
  // mlp hidden = gelu(h2 @ w_mlp1^T + b_mlp1)
  k_gemm_mlp1<<<dim3(4 * D_ / 64, MS_ / 32), dim3(32), 0, stream>>>(h2, wmlp1_bf, b_mlp1, mlp_h);
  // out = x2 + gate_mlp * (mlp_h @ w_mlp2^T + b_mlp2)
  k_gemm_mlp2<<<dim3(D_ / 64, MS_ / 32), dim3(32), 0, stream>>>(mlp_h, wmlp2_bf, b_mlp2, x2, ada, out);
}
